// MLPTexture3D_20057497272423
// MI455X (gfx1250) — compile-verified, matrix operand order checked by reading, not run on hardware
//
#include <hip/hip_runtime.h>
#include <math.h>

typedef __attribute__((ext_vector_type(16))) _Float16 v16h;
typedef __attribute__((ext_vector_type(8)))  float    v8f;

#define NUM_LEVELS   16
#define TABLE_SIZE   (1u << 19)
#define HASH_MASK    (TABLE_SIZE - 1u)
#define BASE_RES     16.0f
#define LEVEL_SCALE  1.4472692374403782f   /* exp(ln(4096/16)/15) */
#define BLOCK_PTS    128                   /* 4 waves x 32 lanes, 1 point/thread */

// Load one 32x16 B-tile of an MLP weight matrix (row-major [out][32]) into the
// 16-bit WMMA B register layout: lane = N column (&15), halves = K per the
// CDNA5 16-bit operand layout (lanes 16-31 hold K offset by 8 in each group).
__device__ __forceinline__ v16h load_B16(const float* __restrict__ W, int n, int laneHi, bool valid)
{
    v16h b;
#pragma unroll
    for (int i = 0; i < 16; ++i) {
        int k = ((i >> 3) << 4) + laneHi * 8 + (i & 7);
        b[i] = valid ? (_Float16)W[n * 32 + k] : (_Float16)0.0f;
    }
    return b;
}

__device__ __forceinline__ v8f wmma_f16(v16h a, v16h b, v8f c)
{
    // D = A(16x32 f16) * B(32x16 f16) + C(16x16 f32)
    return __builtin_amdgcn_wmma_f32_16x16x32_f16(false, a, false, b, (short)0, c, false, false);
}

__global__ __launch_bounds__(BLOCK_PTS)
void hashgrid_mlp_kernel(const float* __restrict__ points,
                         const float* __restrict__ tables,
                         const float* __restrict__ W1,
                         const float* __restrict__ W2,
                         const float* __restrict__ W3,
                         const float* __restrict__ aabb,
                         float* __restrict__ out,
                         int nPoints)
{
    // encA: 8 A-tiles (16 pts x 32 feats, f16, pre-swizzled to WMMA A layout)
    // hbuf: per-wave staging for hidden activations (C-layout -> A-layout)
    __shared__ v16h encA[8 * 32];
    __shared__ v16h hbuf[4 * 32];
    _Float16* encH = (_Float16*)encA;
    _Float16* hH   = (_Float16*)hbuf;

    const int tid       = threadIdx.x;
    const int lane      = tid & 31;
    const int wave      = tid >> 5;
    const int laneHi    = lane >> 4;
    const int nCol      = lane & 15;
    const int blockBase = (int)blockIdx.x * BLOCK_PTS;

    // ---- MLP weights -> WMMA B-layout registers (L1/L2 broadcast hits) ----
    v16h b1lo = load_B16(W1, nCol,      laneHi, true);
    v16h b1hi = load_B16(W1, nCol + 16, laneHi, true);
    v16h b2lo = load_B16(W2, nCol,      laneHi, true);
    v16h b2hi = load_B16(W2, nCol + 16, laneHi, true);
    v16h b3   = load_B16(W3, nCol,      laneHi, nCol < 4);

    // ---- Phase 1: hash-grid encode (one point per thread) ----
    int p  = blockBase + tid;
    int pc = (p < nPoints) ? p : (nPoints - 1);
    float px = points[pc * 3 + 0];
    float py = points[pc * 3 + 1];
    float pz = points[pc * 3 + 2];
    float a0 = aabb[0], a1 = aabb[1], a2 = aabb[2];
    float m0 = aabb[3], m1 = aabb[4], m2 = aabb[5];
    float tx = fminf(fmaxf((px - a0) / (m0 - a0), 0.0f), 1.0f);
    float ty = fminf(fmaxf((py - a1) / (m1 - a1), 0.0f), 1.0f);
    float tz = fminf(fmaxf((pz - a2) / (m2 - a2), 0.0f), 1.0f);

    const int mRow = tid & 15;   // row inside the 16-point A tile
    const int tile = tid >> 4;   // which of the 8 A tiles this point belongs to

    float grow = BASE_RES;
#pragma unroll 1
    for (int l = 0; l < NUM_LEVELS; ++l) {
        float scale = grow - 1.0f;
        grow *= LEVEL_SCALE;
        int      res  = (int)ceilf(scale) + 1;
        unsigned ures = (unsigned)res;
        unsigned rm1  = ures - 1u;
        bool dense = ((unsigned long long)ures * ures * ures) <= (unsigned long long)TABLE_SIZE;
        const float* tab = tables + (size_t)l * (size_t)TABLE_SIZE * 2u;

        float fx = tx * scale + 0.5f;
        float fy = ty * scale + 0.5f;
        float fz = tz * scale + 0.5f;
        float gx = floorf(fx), gy = floorf(fy), gz = floorf(fz);
        float wx = fx - gx, wy = fy - gy, wz = fz - gz;
        unsigned bx = (unsigned)gx, by = (unsigned)gy, bz = (unsigned)gz;

        float acc0 = 0.0f, acc1 = 0.0f;
#pragma unroll
        for (int c = 0; c < 8; ++c) {           // 8 independent L2 gathers in flight
            unsigned ox = c & 1u, oy = (c >> 1) & 1u, oz = (c >> 2) & 1u;
            unsigned cx = bx + ox; cx = cx < rm1 ? cx : rm1;
            unsigned cy = by + oy; cy = cy < rm1 ? cy : rm1;
            unsigned cz = bz + oz; cz = cz < rm1 ? cz : rm1;
            unsigned idx;
            if (dense) idx = cx + cy * ures + cz * ures * ures;
            else       idx = (cx ^ (cy * 2654435761u) ^ (cz * 805459861u)) & HASH_MASK;
            float wt = (ox ? wx : 1.0f - wx) * (oy ? wy : 1.0f - wy) * (oz ? wz : 1.0f - wz);
            float2 tv = *(const float2*)(tab + (size_t)idx * 2u);
            acc0 = fmaf(wt, tv.x, acc0);
            acc1 = fmaf(wt, tv.y, acc1);
        }
        // scatter features f = 2l, 2l+1 directly into WMMA A layout:
        //   target lane = m + 16*bit3(f), element = (f&7) + 8*(f>=16)
#pragma unroll
        for (int j = 0; j < 2; ++j) {
            int f  = 2 * l + j;
            int Lt = mRow + (((f >> 3) & 1) << 4);
            int ii = (f & 7) + ((f >= 16) ? 8 : 0);
            encH[tile * 512 + Lt * 16 + ii] = (_Float16)(j ? acc1 : acc0);
        }
    }
    // wave-local ordering: DS is in-order per wave; drain before cross-lane read
    asm volatile("s_wait_dscnt 0x0" ::: "memory");

    // ---- Phase 2: WMMA MLP, two 16-point tiles per wave, no block barriers ----
#pragma unroll
    for (int tt = 0; tt < 2; ++tt) {
        const int t = wave * 2 + tt;
        v16h a = encA[t * 32 + lane];          // contiguous 32B ds_load per lane
        v8f z = {};
        v8f c0 = wmma_f16(a, b1lo, z);
        v8f c1 = wmma_f16(a, b1hi, z);

        // ReLU on f32 accumulators, redistribute C-layout -> A-layout via LDS
#pragma unroll
        for (int v = 0; v < 8; ++v) {
            int m  = v + laneHi * 8;
            float v0 = fmaxf(c0[v], 0.0f);     // feature f0 = nCol       (0..15)
            float v1 = fmaxf(c1[v], 0.0f);     // feature f1 = nCol + 16  (16..31)
            int f0  = nCol;
            int Lt0 = m + (((f0 >> 3) & 1) << 4);
            hH[wave * 512 + Lt0 * 16 + (f0 & 7)] = (_Float16)v0;
            int f1  = nCol + 16;
            int Lt1 = m + (((f1 >> 3) & 1) << 4);
            hH[wave * 512 + Lt1 * 16 + (f1 & 7) + 8] = (_Float16)v1;
        }
        asm volatile("s_wait_dscnt 0x0" ::: "memory");

        v16h h1 = hbuf[wave * 32 + lane];
        v8f d0 = wmma_f16(h1, b2lo, z);
        v8f d1 = wmma_f16(h1, b2hi, z);
#pragma unroll
        for (int v = 0; v < 8; ++v) {
            int m  = v + laneHi * 8;
            float v0 = fmaxf(d0[v], 0.0f);
            float v1 = fmaxf(d1[v], 0.0f);
            int f0  = nCol;
            int Lt0 = m + (((f0 >> 3) & 1) << 4);
            hH[wave * 512 + Lt0 * 16 + (f0 & 7)] = (_Float16)v0;
            int f1  = nCol + 16;
            int Lt1 = m + (((f1 >> 3) & 1) << 4);
            hH[wave * 512 + Lt1 * 16 + (f1 & 7) + 8] = (_Float16)v1;
        }
        asm volatile("s_wait_dscnt 0x0" ::: "memory");

        v16h h2 = hbuf[wave * 32 + lane];
        v8f o = wmma_f16(h2, b3, z);           // only N<4 columns are valid

        if (nCol < 4) {
            float mn  = (nCol == 3) ? 0.01f : 0.0f;
            float rng = 1.0f - mn;
#pragma unroll
            for (int v = 0; v < 8; ++v) {
                int m  = v + laneHi * 8;
                int pt = blockBase + t * 16 + m;
                if (pt < nPoints) {
                    float s = 1.0f / (1.0f + __expf(-o[v]));
                    out[pt * 4 + nCol] = fmaf(s, rng, mn);
                }
            }
        }
    }
}

extern "C" void kernel_launch(void* const* d_in, const int* in_sizes, int n_in,
                              void* d_out, int out_size, void* d_ws, size_t ws_size,
                              hipStream_t stream)
{
    const float* points = (const float*)d_in[0];
    const float* tables = (const float*)d_in[1];
    const float* W1     = (const float*)d_in[2];
    const float* W2     = (const float*)d_in[3];
    const float* W3     = (const float*)d_in[4];
    const float* aabb   = (const float*)d_in[5];
    float* out = (float*)d_out;

    int nPoints = in_sizes[0] / 3;
    int blocks  = (nPoints + BLOCK_PTS - 1) / BLOCK_PTS;
    hipLaunchKernelGGL(hashgrid_mlp_kernel, dim3(blocks), dim3(BLOCK_PTS), 0, stream,
                       points, tables, W1, W2, W3, aabb, out, nPoints);
    (void)n_in; (void)out_size; (void)d_ws; (void)ws_size;
}
